// MinimalTransformer_15393162788904
// MI455X (gfx1250) — compile-verified, matrix operand order checked by reading, not run on hardware
//
#include <hip/hip_runtime.h>
#include <hip/hip_bf16.h>

// ---------------------------------------------------------------------------
// MI455X (gfx1250) minimal-transformer forward.
// All matmuls run on V_WMMA_F32_16X16X32_F16 (f16 in, f32 accumulate).
// Wave indices pass through readfirstlane so every loop enclosing a WMMA is
// scalar (SGPR) controlled: EXEC stays all-1s as the ISA requires.
// ---------------------------------------------------------------------------

typedef __attribute__((ext_vector_type(16))) _Float16 v16h;
typedef __attribute__((ext_vector_type(8)))  _Float16 v8h;
typedef __attribute__((ext_vector_type(8)))  float    v8f;

// Load one 16x32 f16 fragment (A layout == B layout for K-major rows).
// Lane L<16 : row L,    K = 0..7  and 16..23
// Lane L>=16: row L-16, K = 8..15 and 24..31
static __device__ __forceinline__ v16h frag_load(const _Float16* __restrict__ base, int ld) {
  const int lane = threadIdx.x & 31;
  const _Float16* p = base + (size_t)(lane & 15) * ld + ((lane >> 4) << 3);
  union { v16h v; v8h h[2]; } u;
  u.h[0] = *reinterpret_cast<const v8h*>(p);
  u.h[1] = *reinterpret_cast<const v8h*>(p + 16);
  return u.v;
}

static __device__ __forceinline__ v8f wmma_f16(v16h a, v16h b, v8f c) {
  return __builtin_amdgcn_wmma_f32_16x16x32_f16(false, a, false, b, (short)0, c, false, false);
}

// ---------------------------------------------------------------------------
// Generic GEMM: out[M,N] = A[M,K](f16) * W[N,K]^T(f16) + bias, optional exact
// GELU, optional f32 residual add, writes f32 and/or f16 outputs.
// Block = 128 threads (4 waves), tile 128(M) x 64(N). Wave w owns rows
// [128*by + 32w, +32) x all 64 columns: 8 WMMAs per 6 fragment loads.
// ---------------------------------------------------------------------------
template <bool GELU_ACT>
__global__ void __launch_bounds__(128)
gemm_kernel(const _Float16* __restrict__ A, const _Float16* __restrict__ W,
            const float* __restrict__ bias, const float* __restrict__ res,
            float* __restrict__ outF, _Float16* __restrict__ outH,
            int M, int N, int K) {
  const int wave = __builtin_amdgcn_readfirstlane(threadIdx.x >> 5);  // SGPR
  const int m0 = blockIdx.y * 128 + wave * 32;
  const int n0 = blockIdx.x * 64;
  const _Float16* Arow0 = A + (size_t)m0 * K;
  const _Float16* Arow1 = A + (size_t)(m0 + 16) * K;

  v8f acc[2][4] = {{v8f{}, v8f{}, v8f{}, v8f{}}, {v8f{}, v8f{}, v8f{}, v8f{}}};
  for (int k = 0; k < K; k += 32) {
    __builtin_prefetch(Arow0 + k + 32, 0, 1);  // global_prefetch_b8
    v16h a0 = frag_load(Arow0 + k, K);
    v16h a1 = frag_load(Arow1 + k, K);
#pragma unroll
    for (int j = 0; j < 4; ++j) {
      v16h b = frag_load(W + (size_t)(n0 + j * 16) * K + k, K);
      acc[0][j] = wmma_f16(a0, b, acc[0][j]);
      acc[1][j] = wmma_f16(a1, b, acc[1][j]);
    }
  }

  const int lane = threadIdx.x & 31;
  const int col = lane & 15;
  const int rb = (lane >> 4) << 3;  // rows rb..rb+7 of each 16-row tile
#pragma unroll
  for (int mt = 0; mt < 2; ++mt) {
#pragma unroll
    for (int j = 0; j < 4; ++j) {
      const int c = n0 + j * 16 + col;
      const float bv = bias[c];
#pragma unroll
      for (int i = 0; i < 8; ++i) {
        const int r = m0 + mt * 16 + rb + i;
        float v = acc[mt][j][i] + bv;
        if (GELU_ACT) v = 0.5f * v * (1.0f + erff(v * 0.70710678118654752f));
        const size_t idx = (size_t)r * N + c;
        if (res)  v += res[idx];
        if (outF) outF[idx] = v;
        if (outH) outH[idx] = (_Float16)v;
      }
    }
  }
}

// ---------------------------------------------------------------------------
// Attention: one block = (b, h, 16-query chunk). Exact two-pass softmax.
// Phase 1: S = scale * Q K^T (WMMA, K-dim = 64) into LDS (causal-masked).
// Phase 2: row softmax in f32, P(f16) into LDS (zero-padded to K%32).
// Phase 3: O = P V via WMMA with Vt[b,h,d,t] (K-dim = sequence).
// LDS strides padded so 16-lane fragment reads hit 64 distinct banks.
// ---------------------------------------------------------------------------
#define SCOL 1028  // f32 row stride (4112 B)
#define PCOL 1032  // f16 row stride (2064 B, 16B aligned, bank-spread)

__global__ void __launch_bounds__(128)
attn_kernel(const _Float16* __restrict__ Q, const _Float16* __restrict__ Kh,
            const _Float16* __restrict__ Vt, _Float16* __restrict__ O) {
  __shared__ float    S[16 * SCOL];
  __shared__ _Float16 P[16 * PCOL];
  __shared__ float    red[128];

  const int T = 1024, HD = 64, Hn = 4, D = 256;
  const int qc = blockIdx.x;  // T/16 chunks
  const int h  = blockIdx.y;
  const int b  = blockIdx.z;
  const int q0 = qc * 16;
  const int wave = __builtin_amdgcn_readfirstlane(threadIdx.x >> 5);  // SGPR
  const int lane = threadIdx.x & 31;
  const int col = lane & 15;
  const int rb  = (lane >> 4) << 3;

  const _Float16* Qb = Q  + ((size_t)(b * Hn + h) * T + q0) * HD;
  const _Float16* Kb = Kh + (size_t)(b * Hn + h) * T * HD;
  const _Float16* Vb = Vt + (size_t)(b * Hn + h) * HD * T;

  const int ntiles = qc + 1;          // causal: only key tiles <= query tile
  const float scale = 0.125f;         // 1/sqrt(64)

  // ---- Phase 1: scores (scalar loop: wave is in an SGPR) ----
  v16h qa0 = frag_load(Qb + 0, HD);
  v16h qa1 = frag_load(Qb + 32, HD);
  for (int j = wave; j < ntiles; j += 4) {
    v8f c = {};
    c = wmma_f16(qa0, frag_load(Kb + (size_t)j * 16 * HD + 0, HD), c);
    c = wmma_f16(qa1, frag_load(Kb + (size_t)j * 16 * HD + 32, HD), c);
    const int cg = j * 16 + col;
#pragma unroll
    for (int i = 0; i < 8; ++i) {
      const int r = rb + i;
      S[r * SCOL + cg] = (cg <= q0 + r) ? c[i] * scale : -3.0e38f;
    }
  }
  __syncthreads();

  // ---- Phase 2: softmax (8 threads per row) ----
  const int send = ntiles * 16;
  const int send32 = (send + 31) & ~31;
  const int row = threadIdx.x >> 3;
  const int sub = threadIdx.x & 7;

  float m = -3.0e38f;
  for (int c = sub; c < send; c += 8) m = fmaxf(m, S[row * SCOL + c]);
  red[threadIdx.x] = m;
  __syncthreads();
  float mrow = red[row * 8];
#pragma unroll
  for (int t = 1; t < 8; ++t) mrow = fmaxf(mrow, red[row * 8 + t]);
  __syncthreads();

  float sacc = 0.f;
  for (int c = sub; c < send; c += 8) {
    float e = __expf(S[row * SCOL + c] - mrow);  // masked entries -> 0
    S[row * SCOL + c] = e;
    sacc += e;
  }
  red[threadIdx.x] = sacc;
  __syncthreads();
  float l = 0.f;
#pragma unroll
  for (int t = 0; t < 8; ++t) l += red[row * 8 + t];
  const float inv = 1.0f / l;
  for (int c = sub; c < send32; c += 8) {
    float e = (c < send) ? S[row * SCOL + c] * inv : 0.f;
    P[row * PCOL + c] = (_Float16)e;
  }
  __syncthreads();

  // ---- Phase 3: O = P V, wave owns d-tile [16w, 16w+16) ----
  const int d0 = wave * 16;
  v8f oc = {};
  for (int s = 0; s < send32; s += 32) {
    v16h pa = frag_load((const _Float16*)P + s, PCOL);
    v16h vb = frag_load(Vb + (size_t)d0 * T + s, T);
    oc = wmma_f16(pa, vb, oc);
  }
#pragma unroll
  for (int i = 0; i < 8; ++i) {
    const int r = q0 + rb + i;
    O[((size_t)b * T + r) * D + h * HD + d0 + col] = (_Float16)oc[i];
  }
}

// ---------------------------------------------------------------------------
// Elementwise helpers
// ---------------------------------------------------------------------------
__global__ void embed_kernel(const int* __restrict__ tokens, const float* __restrict__ tok_emb,
                             const float* __restrict__ pos_emb, float* __restrict__ xf,
                             _Float16* __restrict__ xh) {
  const int idx = blockIdx.x * blockDim.x + threadIdx.x;  // over B*T*256
  const int d = idx & 255;
  const int bt = idx >> 8;
  const int t = bt & 1023;
  const int tok = tokens[bt];
  const float v = tok_emb[(size_t)tok * 256 + d] + pos_emb[(size_t)t * 256 + d];
  xf[idx] = v;
  xh[idx] = (_Float16)v;
}

// qkv[bt, 768] -> Q[b,h,t,d], K[b,h,t,d], Vt[b,h,d,t]  (f16, head-major)
__global__ void qkv_split_kernel(const _Float16* __restrict__ qkv, _Float16* __restrict__ Qh,
                                 _Float16* __restrict__ Kh, _Float16* __restrict__ Vt) {
  const int idx = blockIdx.x * blockDim.x + threadIdx.x;  // over B*T*256
  const int j = idx & 255;
  const int bt = idx >> 8;
  const int b = bt >> 10;
  const int t = bt & 1023;
  const int h = j >> 6;
  const int d = j & 63;
  const _Float16* src = qkv + (size_t)bt * 768;
  const size_t qi = ((size_t)(b * 4 + h) * 1024 + t) * 64 + d;
  Qh[qi] = src[j];
  Kh[qi] = src[256 + j];
  Vt[((size_t)(b * 4 + h) * 64 + d) * 1024 + t] = src[512 + j];
}

__global__ void cvt_f32_f16_kernel(const float* __restrict__ s, _Float16* __restrict__ d, int n) {
  const int i = blockIdx.x * blockDim.x + threadIdx.x;
  if (i < n) d[i] = (_Float16)s[i];
}

// ---------------------------------------------------------------------------
// Host orchestration
// ---------------------------------------------------------------------------
extern "C" void kernel_launch(void* const* d_in, const int* in_sizes, int n_in,
                              void* d_out, int out_size, void* d_ws, size_t ws_size,
                              hipStream_t stream) {
  (void)in_sizes; (void)n_in; (void)out_size; (void)ws_size;
  const int B = 8, T = 1024, D = 256, H = 4, L = 3, DFF = 1024, VOC = 8000;
  const int M = B * T;  // 8192 token rows

  const int*   tokens     = (const int*)d_in[0];
  const float* tok_emb    = (const float*)d_in[1];
  const float* pos_emb    = (const float*)d_in[2];
  const float* in_proj_w  = (const float*)d_in[3];
  const float* in_proj_b  = (const float*)d_in[4];
  const float* attn_out_w = (const float*)d_in[5];
  const float* attn_out_b = (const float*)d_in[6];
  const float* mlp_w1     = (const float*)d_in[7];
  const float* mlp_b1     = (const float*)d_in[8];
  const float* mlp_w2     = (const float*)d_in[9];
  const float* mlp_b2     = (const float*)d_in[10];
  const float* out_w      = (const float*)d_in[11];
  const float* out_b      = (const float*)d_in[12];
  float* logits = (float*)d_out;

  // Workspace carve-out (~68 MB total)
  char* wp = (char*)d_ws;
  auto alloc = [&](size_t bytes) -> char* {
    char* p = wp;
    wp += (bytes + 255) & ~(size_t)255;
    return p;
  };
  float*    x32   = (float*)   alloc((size_t)M * D * 4);
  _Float16* x16   = (_Float16*)alloc((size_t)M * D * 2);
  _Float16* qkv16 = (_Float16*)alloc((size_t)M * 3 * D * 2);
  _Float16* q16   = (_Float16*)alloc((size_t)M * D * 2);
  _Float16* k16   = (_Float16*)alloc((size_t)M * D * 2);
  _Float16* vt16  = (_Float16*)alloc((size_t)M * D * 2);
  _Float16* o16   = (_Float16*)alloc((size_t)M * D * 2);
  _Float16* h16   = (_Float16*)alloc((size_t)M * DFF * 2);
  _Float16* inw16 = (_Float16*)alloc((size_t)L * 3 * D * D * 2);
  _Float16* aow16 = (_Float16*)alloc((size_t)L * D * D * 2);
  _Float16* w116  = (_Float16*)alloc((size_t)L * DFF * D * 2);
  _Float16* w216  = (_Float16*)alloc((size_t)L * D * DFF * 2);
  _Float16* ow16  = (_Float16*)alloc((size_t)VOC * D * 2);

  auto cvt = [&](const float* s, _Float16* dd, int n) {
    cvt_f32_f16_kernel<<<(n + 255) / 256, 256, 0, stream>>>(s, dd, n);
  };
  cvt(in_proj_w,  inw16, L * 3 * D * D);
  cvt(attn_out_w, aow16, L * D * D);
  cvt(mlp_w1,     w116,  L * DFF * D);
  cvt(mlp_w2,     w216,  L * D * DFF);
  cvt(out_w,      ow16,  VOC * D);

  embed_kernel<<<M * D / 256, 256, 0, stream>>>(tokens, tok_emb, pos_emb, x32, x16);

  auto gemm = [&](const _Float16* A, const _Float16* Wm, const float* bias,
                  const float* res, float* oF, _Float16* oH,
                  int m, int n, int k, bool gelu) {
    dim3 grid(n / 64, m / 128);
    if (gelu)
      gemm_kernel<true><<<grid, 128, 0, stream>>>(A, Wm, bias, res, oF, oH, m, n, k);
    else
      gemm_kernel<false><<<grid, 128, 0, stream>>>(A, Wm, bias, res, oF, oH, m, n, k);
  };

  for (int l = 0; l < L; ++l) {
    // QKV projection: [8192,256] x [768,256]^T -> f16
    gemm(x16, inw16 + (size_t)l * 3 * D * D, in_proj_b + (size_t)l * 3 * D,
         nullptr, nullptr, qkv16, M, 3 * D, D, false);
    qkv_split_kernel<<<M * D / 256, 256, 0, stream>>>(qkv16, q16, k16, vt16);
    // Causal attention per (b,h,16-row query chunk)
    attn_kernel<<<dim3(T / 16, H, B), 128, 0, stream>>>(q16, k16, vt16, o16);
    // Output projection + residual (in-place into x32/x16)
    gemm(o16, aow16 + (size_t)l * D * D, attn_out_b + (size_t)l * D,
         x32, x32, x16, M, D, D, false);
    // MLP up + exact GELU
    gemm(x16, w116 + (size_t)l * DFF * D, mlp_b1 + (size_t)l * DFF,
         nullptr, nullptr, h16, M, DFF, D, true);
    // MLP down + residual
    gemm(h16, w216 + (size_t)l * D * DFF, mlp_b2 + (size_t)l * D,
         x32, x32, x16, M, D, DFF, false);
  }
  // Logits: [8192,256] x [8000,256]^T -> f32 output
  gemm(x16, ow16, out_b, nullptr, logits, nullptr, M, VOC, D, false);
}
